// LSTMDecoder_62113817035540
// MI455X (gfx1250) — matrix operand order repaired and checked by reading, not compiled, not run on hardware
//
#include <hip/hip_runtime.h>
#include <hip/hip_bf16.h>

#define N_NODES 131072
#define BATCH   64
#define HENC    256
#define HDEC    256
#define VOCAB   8000
#define MAXLEN  16

typedef __attribute__((ext_vector_type(16))) __bf16 v16bf;
typedef __attribute__((ext_vector_type(8)))  float  v8f;

// native RNE conversion -> v_cvt_*bf16_f32 hardware ops (instead of integer RNE emulation)
__device__ __forceinline__ unsigned short f2bf(float f) {
  __bf16 b = (__bf16)f;
  return __builtin_bit_cast(unsigned short, b);
}
__device__ __forceinline__ float bf2f(unsigned short s) {
  return __uint_as_float(((unsigned)s) << 16);
}

union Frag { v16bf v; unsigned short u[16]; __bf16 h[16]; uint4 q[2]; };

// ---- WMMA fragment loads (ISA 7.12.2 layouts, wave32) ----

// B fragment (K x 16), B[k][n] = W[n][k], W row-major [Nout, K] in bf16.
// lane: n = n0 + (lane&15); chunks at k0 + hi and k0 + 16 + hi, hi = lane<16?0:8.
__device__ __forceinline__ v16bf load_B(const unsigned short* W, int n0, int k0, int ldK) {
  int lane = threadIdx.x & 31;
  const unsigned short* p = W + (size_t)(n0 + (lane & 15)) * ldK + k0 + ((lane & 16) ? 8 : 0);
  Frag f;
  f.q[0] = ((const uint4*)p)[0];
  f.q[1] = ((const uint4*)(p + 16))[0];
  return f.v;
}

// A fragment (16 x 32) from bf16 row-major [M, K].
// lane: m = m0 + (lane&15); chunks at k0 + hi and k0 + 16 + hi, hi = lane<16?0:8.
__device__ __forceinline__ v16bf load_A_bf16(const unsigned short* X, int m0, int k0, int ldK) {
  int lane = threadIdx.x & 31;
  const unsigned short* p = X + (size_t)(m0 + (lane & 15)) * ldK + k0 + ((lane & 16) ? 8 : 0);
  Frag f;
  f.q[0] = *(const uint4*)(p);
  f.q[1] = *(const uint4*)(p + 16);
  return f.v;
}

// A fragment converted on the fly from fp32 row-major [M, K].
__device__ __forceinline__ v16bf load_A_f32(const float* X, int m0, int k0, int ldK) {
  int lane = threadIdx.x & 31;
  const float* p = X + (size_t)(m0 + (lane & 15)) * ldK + k0 + ((lane & 16) ? 8 : 0);
  float4 a0 = ((const float4*)p)[0];
  float4 a1 = ((const float4*)p)[1];
  float4 b0 = ((const float4*)(p + 16))[0];
  float4 b1 = ((const float4*)(p + 16))[1];
  Frag f;
  f.h[0]=(__bf16)a0.x; f.h[1]=(__bf16)a0.y; f.h[2]=(__bf16)a0.z; f.h[3]=(__bf16)a0.w;
  f.h[4]=(__bf16)a1.x; f.h[5]=(__bf16)a1.y; f.h[6]=(__bf16)a1.z; f.h[7]=(__bf16)a1.w;
  f.h[8]=(__bf16)b0.x; f.h[9]=(__bf16)b0.y; f.h[10]=(__bf16)b0.z; f.h[11]=(__bf16)b0.w;
  f.h[12]=(__bf16)b1.x; f.h[13]=(__bf16)b1.y; f.h[14]=(__bf16)b1.z; f.h[15]=(__bf16)b1.w;
  return f.v;
}

__device__ __forceinline__ v8f wmma_bf16(v16bf a, v16bf b, v8f c) {
  return __builtin_amdgcn_wmma_f32_16x16x32_bf16(false, a, false, b, (short)0, c, false, false);
}

// -------- one-time conversion kernels --------
__global__ void k_cvt(const float* __restrict__ src, unsigned short* __restrict__ dst, int n) {
  int i = blockIdx.x * blockDim.x + threadIdx.x;
  if (i < n) dst[i] = f2bf(src[i]);
}
// W_attn [K=256][N=256] row-major -> dst[n][k] (transposed) so B loads are contiguous
__global__ void k_cvtT(const float* __restrict__ src, unsigned short* __restrict__ dst) {
  int i = blockIdx.x * blockDim.x + threadIdx.x;
  int n = i >> 8, k = i & 255;
  dst[n * 256 + k] = f2bf(src[k * 256 + n]);
}
__global__ void k_zero(float* p, int n) {
  int i = blockIdx.x * blockDim.x + threadIdx.x;
  if (i < n) p[i] = 0.f;
}
__global__ void k_copy_gt(const long long* __restrict__ gt, long long* __restrict__ dst, int n) {
  int i = blockIdx.x * blockDim.x + threadIdx.x;
  if (i < n) dst[i] = gt[i];
}

// -------- init: h0/c0 from roots, cur = gt[0], bf16 h --------
__global__ void k_init(const float* __restrict__ nh, const float* __restrict__ nm,
                       const long long* __restrict__ roots, const long long* __restrict__ gt0,
                       float* h, float* c, unsigned short* hb, int* cur) {
  int idx = blockIdx.x * blockDim.x + threadIdx.x;   // 64*256
  int b = idx >> 8, f = idx & 255;
  long long r = roots[b];
  float hv = nh[r * HENC + f], cv = nm[r * HENC + f];
  h[idx] = hv; c[idx] = cv; hb[idx] = f2bf(hv);
  if (f == 0) cur[b] = (int)gt0[b];
}

// -------- proj = node_hidden @ W_attn (WMMA, bf16 result) --------
// one wave computes a 16x64 strip (4 N-tiles) reusing the A fragment
__global__ void k_proj(const float* __restrict__ nh, const unsigned short* __restrict__ WattnT,
                       unsigned short* __restrict__ proj) {
  int wg = blockIdx.x * 8 + (threadIdx.x >> 5);   // 32768 waves
  int mtile = wg >> 2, ng = wg & 3;
  int m0 = mtile * 16;
  v8f acc[4] = {};
  for (int k0 = 0; k0 < HENC; k0 += 32) {
    v16bf a = load_A_f32(nh, m0, k0, HENC);
#pragma unroll
    for (int j = 0; j < 4; ++j) {
      v16bf b = load_B(WattnT, (ng * 4 + j) * 16, k0, HENC);
      acc[j] = wmma_bf16(a, b, acc[j]);
    }
  }
  int lane = threadIdx.x & 31;
  int ro = (lane & 16) ? 8 : 0, col = lane & 15;
#pragma unroll
  for (int j = 0; j < 4; ++j) {
    int n = (ng * 4 + j) * 16 + col;
#pragma unroll
    for (int r = 0; r < 8; ++r)
      proj[(size_t)(m0 + r + ro) * HENC + n] = f2bf(acc[j][r]);
  }
}

// -------- scores[n] = dot(proj[n], h[seg[n]]) : one wave per row --------
__global__ void k_scores(const unsigned short* __restrict__ proj, const float* __restrict__ h,
                         const int* __restrict__ segid, float* __restrict__ scores) {
  int n = blockIdx.x * 8 + (threadIdx.x >> 5);
  int lane = threadIdx.x & 31;
  int b = segid[n];
  Frag f; f.q[0] = *(const uint4*)(proj + (size_t)n * HENC + lane * 8);
  const float* hr = h + b * HENC + lane * 8;
  float4 h0 = ((const float4*)hr)[0], h1 = ((const float4*)hr)[1];
  float s = bf2f(f.u[0]) * h0.x + bf2f(f.u[1]) * h0.y + bf2f(f.u[2]) * h0.z + bf2f(f.u[3]) * h0.w
          + bf2f(f.u[4]) * h1.x + bf2f(f.u[5]) * h1.y + bf2f(f.u[6]) * h1.z + bf2f(f.u[7]) * h1.w;
#pragma unroll
  for (int o = 16; o; o >>= 1) s += __shfl_down(s, o, 32);
  if (lane == 0) scores[n] = s;
}

// -------- segment softmax + weighted sum; grid (64 segments, 8 feature slices) --------
__global__ void k_attend(const float* __restrict__ scores, const float* __restrict__ nh,
                         const long long* __restrict__ roots, float* __restrict__ attended) {
  int b = blockIdx.x, slice = blockIdx.y;
  int tid = threadIdx.x;
  int fl = tid & 31, nlane = tid >> 5;            // 8 node-lanes x 32 features
  int f = slice * 32 + fl;
  long long start = roots[b];
  long long end = (b == BATCH - 1) ? (long long)N_NODES : roots[b + 1];
  __shared__ float sm[256];
  __shared__ float es[256];
  // pass 1: segment max
  float m = -3.402823466e38f;
  for (long long i = start + tid; i < end; i += 256) m = fmaxf(m, scores[i]);
  sm[tid] = m; __syncthreads();
  for (int o = 128; o; o >>= 1) { if (tid < o) sm[tid] = fmaxf(sm[tid], sm[tid + o]); __syncthreads(); }
  float smax = sm[0]; __syncthreads();
  // pass 2: exp, z, weighted accumulation (chunked through LDS)
  float zp = 0.f, att = 0.f;
  for (long long c0 = start; c0 < end; c0 += 256) {
    long long i = c0 + tid;
    float e = (i < end) ? expf(scores[i] - smax) : 0.f;
    es[tid] = e; zp += e;
    __syncthreads();
    long long rem = end - c0;
    int cnt = rem < 256 ? (int)rem : 256;
    for (int j = nlane; j < cnt; j += 8)
      att += es[j] * nh[(c0 + j) * HENC + f];
    __syncthreads();
  }
  sm[tid] = zp; __syncthreads();
  for (int o = 128; o; o >>= 1) { if (tid < o) sm[tid] += sm[tid + o]; __syncthreads(); }
  float z = sm[0]; __syncthreads();
  es[tid] = att; __syncthreads();
  if (nlane == 0) {
    float s = 0.f;
#pragma unroll
    for (int k = 0; k < 8; ++k) s += es[k * 32 + fl];
    attended[b * HENC + f] = s / z;
  }
}

// -------- x = concat(embedding[cur], attended) in bf16 --------
__global__ void k_prep(const float* __restrict__ emb, const int* __restrict__ cur,
                       const float* __restrict__ attended, unsigned short* __restrict__ xb) {
  int idx = blockIdx.x * blockDim.x + threadIdx.x;   // 64*512
  int b = idx >> 9, j = idx & 511;
  float v = (j < 256) ? emb[(size_t)cur[b] * HDEC + j] : attended[b * HENC + (j - 256)];
  xb[idx] = f2bf(v);
}

// -------- gates = x@W_ih^T + h@W_hh^T + biases (WMMA) --------
__global__ void k_gates(const unsigned short* __restrict__ xb, const unsigned short* __restrict__ hb,
                        const unsigned short* __restrict__ Wih, const unsigned short* __restrict__ Whh,
                        const float* __restrict__ bih, const float* __restrict__ bhh,
                        float* __restrict__ gates) {
  int wg = blockIdx.x * 4 + (threadIdx.x >> 5);   // 256 waves
  int mt = wg & 3, nt = wg >> 2;
  int m0 = mt * 16, n0 = nt * 16;
  v8f acc = {};
  for (int k0 = 0; k0 < 512; k0 += 32)
    acc = wmma_bf16(load_A_bf16(xb, m0, k0, 512), load_B(Wih, n0, k0, 512), acc);
  for (int k0 = 0; k0 < 256; k0 += 32)
    acc = wmma_bf16(load_A_bf16(hb, m0, k0, 256), load_B(Whh, n0, k0, 256), acc);
  int lane = threadIdx.x & 31;
  int ro = (lane & 16) ? 8 : 0, n = n0 + (lane & 15);
  float bias = bih[n] + bhh[n];
#pragma unroll
  for (int r = 0; r < 8; ++r)
    gates[(m0 + r + ro) * 1024 + n] = acc[r] + bias;
}

// -------- LSTM elementwise --------
__global__ void k_lstm(const float* __restrict__ gates, float* c, float* h, unsigned short* hb) {
  int idx = blockIdx.x * blockDim.x + threadIdx.x;   // 64*256
  int b = idx >> 8, f = idx & 255;
  float gi = gates[b * 1024 + f];
  float gf = gates[b * 1024 + 256 + f];
  float gg = gates[b * 1024 + 512 + f];
  float go = gates[b * 1024 + 768 + f];
  float si = 1.f / (1.f + expf(-gi));
  float sf = 1.f / (1.f + expf(-gf));
  float so = 1.f / (1.f + expf(-go));
  float cn = sf * c[idx] + si * tanhf(gg);
  float hn = so * tanhf(cn);
  c[idx] = cn; h[idx] = hn; hb[idx] = f2bf(hn);
}

// -------- logits = h@W_out^T + b_out (WMMA), straight into d_out --------
__global__ void k_logits(const unsigned short* __restrict__ hb, const unsigned short* __restrict__ Wout,
                         const float* __restrict__ bout, float* __restrict__ out) {
  int nt = blockIdx.x, mt = threadIdx.x >> 5;
  int m0 = mt * 16, n0 = nt * 16;
  v8f acc = {};
  for (int k0 = 0; k0 < 256; k0 += 32)
    acc = wmma_bf16(load_A_bf16(hb, m0, k0, 256), load_B(Wout, n0, k0, 256), acc);
  int lane = threadIdx.x & 31;
  int ro = (lane & 16) ? 8 : 0, n = n0 + (lane & 15);
  float bias = bout[n];
#pragma unroll
  for (int r = 0; r < 8; ++r)
    out[(size_t)(m0 + r + ro) * VOCAB + n] = acc[r] + bias;
}

// -------- argmax per batch row (first-index tie-break like jnp.argmax) --------
__global__ void k_argmax(const float* __restrict__ logits, int* __restrict__ cur) {
  int b = blockIdx.x;
  __shared__ float sv[256];
  __shared__ int si[256];
  float bv = -3.402823466e38f; int bi = VOCAB;
  for (int v = threadIdx.x; v < VOCAB; v += 256) {
    float x = logits[(size_t)b * VOCAB + v];
    if (x > bv || (x == bv && v < bi)) { bv = x; bi = v; }
  }
  sv[threadIdx.x] = bv; si[threadIdx.x] = bi; __syncthreads();
  for (int o = 128; o; o >>= 1) {
    if (threadIdx.x < o) {
      float ov = sv[threadIdx.x + o]; int oi = si[threadIdx.x + o];
      if (ov > sv[threadIdx.x] || (ov == sv[threadIdx.x] && oi < si[threadIdx.x])) {
        sv[threadIdx.x] = ov; si[threadIdx.x] = oi;
      }
    }
    __syncthreads();
  }
  if (threadIdx.x == 0) cur[b] = si[0];
}

extern "C" void kernel_launch(void* const* d_in, const int* in_sizes, int n_in,
                              void* d_out, int out_size, void* d_ws, size_t ws_size,
                              hipStream_t stream) {
  const float*      nh    = (const float*)d_in[0];
  const float*      nm    = (const float*)d_in[1];
  const float*      emb   = (const float*)d_in[2];
  const float*      Wattn = (const float*)d_in[3];
  const float*      Wih   = (const float*)d_in[4];
  const float*      Whh   = (const float*)d_in[5];
  const float*      bih   = (const float*)d_in[6];
  const float*      bhh   = (const float*)d_in[7];
  const float*      Wout  = (const float*)d_in[8];
  const float*      bout  = (const float*)d_in[9];
  const long long*  roots = (const long long*)d_in[10];
  const int*        segid = (const int*)d_in[11];
  const long long*  gt    = (const long long*)d_in[12];
  float* out = (float*)d_out;

  // workspace carve-out (~74 MB)
  char* w = (char*)d_ws; size_t off = 0;
  auto carve = [&](size_t bytes) { void* p = w + off; off = (off + bytes + 255) & ~(size_t)255; return p; };
  unsigned short* proj_b  = (unsigned short*)carve((size_t)N_NODES * HENC * 2);
  unsigned short* Wih_b   = (unsigned short*)carve((size_t)1024 * 512 * 2);
  unsigned short* Whh_b   = (unsigned short*)carve((size_t)1024 * 256 * 2);
  unsigned short* Wout_b  = (unsigned short*)carve((size_t)VOCAB * 256 * 2);
  unsigned short* WattnT_b= (unsigned short*)carve((size_t)256 * 256 * 2);
  float*          scores  = (float*)carve((size_t)N_NODES * 4);
  float*          h       = (float*)carve((size_t)BATCH * HENC * 4);
  float*          c       = (float*)carve((size_t)BATCH * HENC * 4);
  unsigned short* hb      = (unsigned short*)carve((size_t)BATCH * HENC * 2);
  unsigned short* xb      = (unsigned short*)carve((size_t)BATCH * 512 * 2);
  float*          attended= (float*)carve((size_t)BATCH * HENC * 4);
  float*          gates   = (float*)carve((size_t)BATCH * 1024 * 4);
  int*            cur     = (int*)carve((size_t)BATCH * 4);

  // one-time conversions
  k_cvt <<<(1024 * 512 + 255) / 256, 256, 0, stream>>>(Wih, Wih_b, 1024 * 512);
  k_cvt <<<(1024 * 256 + 255) / 256, 256, 0, stream>>>(Whh, Whh_b, 1024 * 256);
  k_cvt <<<(VOCAB * 256 + 255) / 256, 256, 0, stream>>>(Wout, Wout_b, VOCAB * 256);
  k_cvtT<<<(256 * 256) / 256, 256, 0, stream>>>(Wattn, WattnT_b);

  // init state, zero t=0 logits, copy ground_truth to tail of d_out
  k_init<<<(BATCH * HENC) / 256, 256, 0, stream>>>(nh, nm, roots, gt, h, c, hb, cur);
  k_zero<<<(BATCH * VOCAB + 255) / 256, 256, 0, stream>>>(out, BATCH * VOCAB);
  k_copy_gt<<<(MAXLEN * BATCH + 255) / 256, 256, 0, stream>>>(
      gt, (long long*)(out + (size_t)MAXLEN * BATCH * VOCAB), MAXLEN * BATCH);

  // loop-invariant projection (WMMA)
  k_proj<<<(N_NODES / 16) * 4 / 8, 256, 0, stream>>>(nh, WattnT_b, proj_b);

  for (int t = 1; t < MAXLEN; ++t) {
    k_scores<<<N_NODES / 8, 256, 0, stream>>>(proj_b, h, segid, scores);
    k_attend<<<dim3(BATCH, 8), 256, 0, stream>>>(scores, nh, roots, attended);
    k_prep  <<<(BATCH * 512) / 256, 256, 0, stream>>>(emb, cur, attended, xb);
    k_gates <<<64, 128, 0, stream>>>(xb, hb, Wih_b, Whh_b, bih, bhh, gates);
    k_lstm  <<<(BATCH * HENC) / 256, 256, 0, stream>>>(gates, c, h, hb);
    float* out_t = out + (size_t)t * BATCH * VOCAB;
    k_logits<<<VOCAB / 16, 128, 0, stream>>>(hb, Wout_b, bout, out_t);
    k_argmax<<<BATCH, 256, 0, stream>>>(out_t, cur);
  }
}
